// RenderNet_71159018160131
// MI455X (gfx1250) — compile-verified, matrix operand order checked by reading, not run on hardware
//
#include <hip/hip_runtime.h>
#include <hip/hip_bf16.h>

typedef __attribute__((ext_vector_type(16))) _Float16 v16h;
typedef __attribute__((ext_vector_type(8)))  _Float16 v8h;
typedef __attribute__((ext_vector_type(8)))  float    v8f;

#define NQ      16384      // 512 rays * 32 samples
#define NPART   4096
#define CAT4_K  480        // 454 padded to multiple of 32
#define DIR_K   320        // 310 padded
#define L0_K    224        // 198 padded

// ---------------------------------------------------------------------------
// Pack fp32 weights [din x dout] into fp16 WMMA-B fragment-major layout.
// Packed index p = (((ntile*KS + ks)*32 + lane)*16 + e), where for element e:
//   k = ks*32 + (e>>3)*16 + (lane>>4)*8 + (e&7),  n = ntile*16 + (lane&15)
// (matches the CDNA5 16-bit 32x16 B-matrix VGPR layout; pad region = 0).
// ---------------------------------------------------------------------------
__global__ void __launch_bounds__(256)
pack_weights(const float* __restrict__ W, _Float16* __restrict__ Wp,
             int din, int dout, int KP, int NP)
{
    const int KS = KP >> 5;
    const int total = KP * NP;
    for (int p = blockIdx.x * 256 + threadIdx.x; p < total; p += gridDim.x * 256) {
        int e    = p & 15;
        int lane = (p >> 4) & 31;
        int rest = p >> 9;
        int ks   = rest % KS;
        int t    = rest / KS;
        int k = (ks << 5) + ((e >> 3) << 4) + ((lane >> 4) << 3) + (e & 7);
        int n = (t << 4) + (lane & 15);
        float v = (k < din && n < dout) ? W[(size_t)k * dout + n] : 0.0f;
        Wp[p] = (_Float16)v;
    }
}

// ---------------------------------------------------------------------------
// Positional encodings (fp16 output)
// ---------------------------------------------------------------------------
__device__ __forceinline__ void posenc3(const float* x, int F, _Float16* dst)
{
    dst[0] = (_Float16)x[0]; dst[1] = (_Float16)x[1]; dst[2] = (_Float16)x[2];
    float s = 1.0f;
    for (int f = 0; f < F; ++f) {
        int b = 3 + f * 6;
        #pragma unroll
        for (int c = 0; c < 3; ++c) {
            float sv, cv; __sincosf(x[c] * s, &sv, &cv);
            dst[b + c]     = (_Float16)sv;
            dst[b + 3 + c] = (_Float16)cv;
        }
        s *= 2.0f;
    }
}

__device__ __forceinline__ void posenc1(float x, int F, _Float16* dst)
{
    dst[0] = (_Float16)x;
    float s = 1.0f;
    for (int f = 0; f < F; ++f) {
        float sv, cv; __sincosf(x * s, &sv, &cv);
        dst[1 + 2 * f] = (_Float16)sv;
        dst[2 + 2 * f] = (_Float16)cv;
        s *= 2.0f;
    }
}

// ---------------------------------------------------------------------------
// Ball query + feature construction. One thread per query point.
// Particles staged through LDS in 512-point chunks (index order preserved so
// "first 16 within radius by index" matches the reference top_k semantics).
// Writes:
//   cat4[i, 0:198]   = [pos_emb(63), den_emb(9), sp_emb(63), var_emb(63)]
//   cat4[i, 198:480] = 0   (h256 written later by layer 3; tail pad stays 0)
//   dircat[i,256:310]= [ray_emb(27), sdir_emb(27)],  dircat[i,310:320]=0
//   mask[i]          = all 16 slots valid with nonzero d2
// ---------------------------------------------------------------------------
__global__ void __launch_bounds__(128)
features_kernel(const float* __restrict__ rp0, const float* __restrict__ pp,
                const float* __restrict__ ro,  const float* __restrict__ rays,
                _Float16* __restrict__ cat4, _Float16* __restrict__ dircat,
                float* __restrict__ mask)
{
    __shared__ float sp[512 * 3];
    const int i = blockIdx.x * 128 + threadIdx.x;

    const float q0 = rp0[(size_t)i * 3 + 0];
    const float q1 = rp0[(size_t)i * 3 + 1];
    const float q2 = rp0[(size_t)i * 3 + 2];

    int   cnt = 0, num = 0;
    bool  allnz = true;
    float wsum = 0.f, wpx = 0.f, wpy = 0.f, wpz = 0.f;
    float svx = 0.f, svy = 0.f, svz = 0.f;
    float qvx = 0.f, qvy = 0.f, qvz = 0.f;   // sum of squares of (p - q)

    for (int c = 0; c < NPART / 512; ++c) {
        __syncthreads();
        for (int j = threadIdx.x; j < 512 * 3; j += 128)
            sp[j] = pp[c * 512 * 3 + j];
        __syncthreads();
        for (int j = 0; j < 512; ++j) {
            float px = sp[j * 3 + 0], py = sp[j * 3 + 1], pz = sp[j * 3 + 2];
            float dx = px - q0, dy = py - q1, dz = pz - q2;
            float d2 = dx * dx + dy * dy + dz * dz;
            if (d2 < 4.0f && cnt < 16) {
                ++cnt;
                float de = sqrtf(d2);
                float w  = fmaxf(1.0f - de * de * de * 0.125f, 0.0f);
                wsum += w; wpx += w * px; wpy += w * py; wpz += w * pz;
                bool nz = (d2 != 0.0f);
                allnz = allnz && nz;
                if (nz) {
                    ++num;
                    svx += dx; svy += dy; svz += dz;
                    qvx += dx * dx; qvy += dy * dy; qvz += dz * dz;
                }
            }
        }
    }

    // empty slots: nbrs = 0 -> de = |q|, contribute wgt only
    float qn = sqrtf(q0 * q0 + q1 * q1 + q2 * q2);
    float w0 = fmaxf(1.0f - qn * qn * qn * 0.125f, 0.0f);
    wsum += (float)(16 - cnt) * w0;

    float density = wsum;
    float invw = 1.0f / (wsum + 1e-12f);
    float spv[3] = { wpx * invw, wpy * invw, wpz * invw };

    float fn   = (float)num;
    float invn = 1.0f / (fn + 1e-12f);
    float vmx = svx * invn, vmy = svy * invn, vmz = svz * invn;
    float var[3] = { (qvx - 2.f * vmx * svx + fn * vmx * vmx) * invn,
                     (qvy - 2.f * vmy * svy + fn * vmy * vmy) * invn,
                     (qvz - 2.f * vmz * svz + fn * vmz * vmz) * invn };

    float sdx = spv[0] - ro[0], sdy = spv[1] - ro[1], sdz = spv[2] - ro[2];
    float sdn = sqrtf(sdx * sdx + sdy * sdy + sdz * sdz);
    float sdir[3] = { sdx / sdn, sdy / sdn, sdz / sdn };

    float qv[3] = { q0, q1, q2 };

    _Float16* c4 = cat4 + (size_t)i * CAT4_K;
    posenc3(qv,  10, c4 + 0);
    posenc1(density, 4, c4 + 63);
    posenc3(spv, 10, c4 + 72);
    posenc3(var, 10, c4 + 135);
    {   // zero cols 198..479 (dword-vectorized; 198 is even)
        unsigned* z = (unsigned*)(c4 + 198);
        #pragma unroll 4
        for (int j = 0; j < (CAT4_K - 198) / 2; ++j) z[j] = 0u;
    }

    _Float16* dc = dircat + (size_t)i * DIR_K;
    const float* ray = rays + (size_t)(i >> 5) * 3;
    float rv[3] = { ray[0], ray[1], ray[2] };
    posenc3(rv,   4, dc + 256);
    posenc3(sdir, 4, dc + 283);
    {
        unsigned* z = (unsigned*)(dc + 310);
        #pragma unroll
        for (int j = 0; j < (DIR_K - 310) / 2; ++j) z[j] = 0u;
    }

    mask[i] = (cnt == 16 && allnz) ? 1.0f : 0.0f;
}

// ---------------------------------------------------------------------------
// WMMA GEMM: C[M x N] = act(A[M x KP] * Wp + bias)
// Block = 256 threads = 8 waves; block owns one 16-row M tile x all N cols.
// A tile staged in LDS once (b128 global->LDS), then each wave processes up
// to TWO 16-wide N tiles simultaneously: one A-fragment DS load feeds two
// independent v_wmma_f32_16x16x32_f16 accumulators per k-step. The wave id
// is forced into an SGPR via readfirstlane so the dual-tile predicate is a
// scalar branch (no EXEC masking around WMMA; EXEC stays all-ones).
// act: 0=none (fp16 out), 1=relu (fp16 out), 2=sigmoid*mask (fp32 rgb out)
// ---------------------------------------------------------------------------
__global__ void __launch_bounds__(256)
gemm_wmma(const _Float16* __restrict__ A, int lda,
          const _Float16* __restrict__ Wp, int KP, int NP,
          const float* __restrict__ bias, int Nreal,
          _Float16* __restrict__ dstH, int ldd, int dstOff,
          float* __restrict__ dstF, const float* __restrict__ mask,
          int act)
{
    __shared__ _Float16 As[16 * CAT4_K];   // max tile: 16 x 480 fp16 = 15 KB

    const int tid = threadIdx.x;
    const int m0  = blockIdx.x * 16;

    // cooperative A-tile load, 128-bit granular (lda is a multiple of 8)
    {
        const uint4* Ag  = (const uint4*)A;
        uint4*       Asw = (uint4*)As;
        const int q4   = KP >> 3;          // uint4 per row
        const int ld4  = lda >> 3;
        const int total = 16 * q4;
        for (int idx = tid; idx < total; idx += 256) {
            int r = idx / q4;
            int c = idx - r * q4;
            Asw[r * q4 + c] = Ag[(size_t)(m0 + r) * ld4 + c];
        }
    }
    __syncthreads();

    const int lane   = tid & 31;
    // wave id is lane-invariant; readfirstlane makes that fact visible to the
    // backend so all tile-selection control flow stays in SGPRs (s_cbranch).
    const int wave   = __builtin_amdgcn_readfirstlane(tid >> 5);
    const int KS     = KP >> 5;
    const int mrow   = lane & 15;
    const int hi8    = (lane >> 4) << 3;   // lane[4] selects K/M halves
    const int ncol   = lane & 15;
    const int ntiles = NP >> 4;

    auto epilogue = [&](int t, const v8f& acc) {
        const int n = t * 16 + ncol;
        if (n < Nreal) {
            const float bn = bias[n];
            #pragma unroll
            for (int r = 0; r < 8; ++r) {
                const int m = m0 + hi8 + r;       // C/D layout: vgpr r -> row
                float v = acc[r] + bn;
                if (act == 1) v = fmaxf(v, 0.0f);
                if (act == 2) {
                    v = (1.0f / (1.0f + __expf(-v))) * mask[m];
                    dstF[(size_t)m * 3 + n] = v;
                } else {
                    dstH[(size_t)m * ldd + dstOff + n] = (_Float16)v;
                }
            }
        }
    };

    for (int t0 = wave; t0 < ntiles; t0 += 16) {
        const int  t1   = t0 + 8;
        const bool dual = (t1 < ntiles);   // scalar (t0, ntiles are SGPRs)
        v8f acc0 = {}, acc1 = {};
        const v16h* bp0 = (const v16h*)Wp + (size_t)t0 * KS * 32 + lane;
        const v16h* bp1 = (const v16h*)Wp + (size_t)t1 * KS * 32 + lane;
        for (int ks = 0; ks < KS; ++ks) {
            // A fragment: lane = row (mrow), two contiguous 8-half K groups
            union { v16h v; v8h h[2]; } a;
            const int base = mrow * KP + (ks << 5) + hi8;
            a.h[0] = *(const v8h*)(As + base);
            a.h[1] = *(const v8h*)(As + base + 16);
            __builtin_prefetch((const void*)(bp0 + (size_t)(ks + 4) * 32), 0, 1);
            v16h b0 = bp0[(size_t)ks * 32];
            acc0 = __builtin_amdgcn_wmma_f32_16x16x32_f16(
                false, a.v, false, b0, (short)0, acc0, false, false);
            if (dual) {
                v16h b1 = bp1[(size_t)ks * 32];
                acc1 = __builtin_amdgcn_wmma_f32_16x16x32_f16(
                    false, a.v, false, b1, (short)0, acc1, false, false);
            }
        }
        epilogue(t0, acc0);
        if (dual) epilogue(t1, acc1);
    }
}

// ---------------------------------------------------------------------------
extern "C" void kernel_launch(void* const* d_in, const int* in_sizes, int n_in,
                              void* d_out, int out_size, void* d_ws, size_t ws_size,
                              hipStream_t stream)
{
    (void)in_sizes; (void)n_in; (void)out_size; (void)ws_size;

    const float* rp0  = (const float*)d_in[0];   // (512,32,3)
    const float* pp   = (const float*)d_in[1];   // (4096,3)
    const float* ro   = (const float*)d_in[2];   // (3,)
    const float* rays = (const float*)d_in[3];   // (512,3)
    // params flat order: xyz_w0,b0 ... xyz_w7,b7, sigma_w,b, final_w,b, dir_w,b, rgb_w,b

    char* ws = (char*)d_ws;
    auto carve = [&](size_t bytes) -> void* {
        void* p = (void*)ws;
        ws += (bytes + 255) & ~(size_t)255;
        return p;
    };

    _Float16* Wx[8];
    Wx[0] = (_Float16*)carve((size_t)L0_K * 256 * 2);
    for (int l = 1; l < 8; ++l)
        Wx[l] = (_Float16*)carve((l == 4 ? (size_t)CAT4_K : 256) * 256 * 2);
    _Float16* Wfin = (_Float16*)carve((size_t)256 * 256 * 2);
    _Float16* Wdir = (_Float16*)carve((size_t)DIR_K * 128 * 2);
    _Float16* Wrgb = (_Float16*)carve((size_t)128 * 16 * 2);

    _Float16* cat4   = (_Float16*)carve((size_t)NQ * CAT4_K * 2);
    _Float16* dircat = (_Float16*)carve((size_t)NQ * DIR_K * 2);
    _Float16* B1     = (_Float16*)carve((size_t)NQ * 256 * 2);
    _Float16* B2     = (_Float16*)carve((size_t)NQ * 256 * 2);
    _Float16* Bdir   = (_Float16*)carve((size_t)NQ * 128 * 2);
    float*    maskb  = (float*)carve((size_t)NQ * 4);

    auto pack = [&](const void* w, _Float16* wp, int din, int dout, int KP, int NP) {
        int total = KP * NP;
        pack_weights<<<(total + 255) / 256, 256, 0, stream>>>(
            (const float*)w, wp, din, dout, KP, NP);
    };
    pack(d_in[4],  Wx[0], 198, 256, L0_K,   256);
    pack(d_in[6],  Wx[1], 256, 256, 256,    256);
    pack(d_in[8],  Wx[2], 256, 256, 256,    256);
    pack(d_in[10], Wx[3], 256, 256, 256,    256);
    pack(d_in[12], Wx[4], 454, 256, CAT4_K, 256);
    pack(d_in[14], Wx[5], 256, 256, 256,    256);
    pack(d_in[16], Wx[6], 256, 256, 256,    256);
    pack(d_in[18], Wx[7], 256, 256, 256,    256);
    pack(d_in[22], Wfin,  256, 256, 256,    256);
    pack(d_in[24], Wdir,  310, 128, DIR_K,  128);
    pack(d_in[26], Wrgb,  128, 3,   128,    16);

    features_kernel<<<NQ / 128, 128, 0, stream>>>(rp0, pp, ro, rays, cat4, dircat, maskb);

    auto gemm = [&](const _Float16* A, int lda, const _Float16* Wp, int KP, int NP,
                    const void* bias, int Nreal,
                    _Float16* dH, int ldd, int off, float* dF, int act) {
        gemm_wmma<<<NQ / 16, 256, 0, stream>>>(A, lda, Wp, KP, NP,
                                               (const float*)bias, Nreal,
                                               dH, ldd, off, dF, maskb, act);
    };

    // xyz MLP (skip concat at layer 4 via cat4 buffer; sigma head unused)
    gemm(cat4, CAT4_K, Wx[0], L0_K,   256, d_in[5],  256, B1,   256,    0,   nullptr, 1);
    gemm(B1,   256,    Wx[1], 256,    256, d_in[7],  256, B2,   256,    0,   nullptr, 1);
    gemm(B2,   256,    Wx[2], 256,    256, d_in[9],  256, B1,   256,    0,   nullptr, 1);
    gemm(B1,   256,    Wx[3], 256,    256, d_in[11], 256, cat4, CAT4_K, 198, nullptr, 1);
    gemm(cat4, CAT4_K, Wx[4], CAT4_K, 256, d_in[13], 256, B1,   256,    0,   nullptr, 1);
    gemm(B1,   256,    Wx[5], 256,    256, d_in[15], 256, B2,   256,    0,   nullptr, 1);
    gemm(B2,   256,    Wx[6], 256,    256, d_in[17], 256, B1,   256,    0,   nullptr, 1);
    gemm(B1,   256,    Wx[7], 256,    256, d_in[19], 256, B2,   256,    0,   nullptr, 1);
    // final (no act) -> dircat[:, 0:256]
    gemm(B2,   256,    Wfin,  256,    256, d_in[23], 256, dircat, DIR_K, 0,  nullptr, 0);
    // dir layer (relu)
    gemm(dircat, DIR_K, Wdir, DIR_K,  128, d_in[25], 128, Bdir, 128,    0,   nullptr, 1);
    // rgb head: sigmoid * mask -> fp32 output (16384 x 3)
    gemm(Bdir, 128,    Wrgb,  128,    16,  d_in[27], 3,   nullptr, 0,   0,   (float*)d_out, 2);
}
